// MHA_34900904247749
// MI455X (gfx1250) — matrix operand order repaired and checked by reading, not compiled, not run on hardware
//
#include <hip/hip_runtime.h>
#include <hip/hip_bf16.h>

// ---------------------------------------------------------------------------
// Feature probes (guarded so the file always compiles on either toolchain)
// ---------------------------------------------------------------------------
#if defined(__gfx1250__)
  #if __has_builtin(__builtin_amdgcn_global_load_async_to_lds_b128)
    #define HAVE_ASYNC 1
  #endif
  #if __has_builtin(__builtin_amdgcn_tensor_load_to_lds)
    #define HAVE_TDM 1
    #if __has_include(<hip/amd_detail/amd_gfx1250_TDM.h>)
      #define TDM_6ARG 1   // amdgpu-toolchain (clang-23): 6-arg builtin
    #endif
  #endif
#endif

// ---------------------------------------------------------------------------
// Types for CDNA5 WMMA (wave32): v_wmma_f32_16x16x32_bf16
// ---------------------------------------------------------------------------
typedef __attribute__((ext_vector_type(16))) __bf16          v16bf;
typedef __attribute__((ext_vector_type(16))) unsigned short  v16u;
typedef __attribute__((ext_vector_type(8)))  float           v8f;
typedef __attribute__((ext_vector_type(4)))  unsigned int    v4u;
typedef __attribute__((ext_vector_type(8)))  int             v8i;
typedef __attribute__((ext_vector_type(4)))  int             v4i;

__device__ __forceinline__ unsigned short f2bf(float f) {
  unsigned int u = __float_as_uint(f);
  unsigned int r = (u + 0x7FFFu + ((u >> 16) & 1u)) >> 16;   // round-nearest-even
  return (unsigned short)r;
}

__device__ __forceinline__ v16bf as_bf16x16(v16u u) {
  return __builtin_bit_cast(v16bf, u);
}

// A-matrix (16x32, 16-bit) fragment: lane holds row (lane%16); element pair
// (2v,2v+1) sits at K = {2v | 16+2(v-4)} + 8*(lane/16).  p -> row start.
__device__ __forceinline__ v16bf frag_a_from(const unsigned short* p, int half) {
  v16u t;
#pragma unroll
  for (int v = 0; v < 8; ++v) {
    int k = ((v < 4) ? 2 * v : 16 + 2 * (v - 4)) + half * 8;
    unsigned int d = *(const unsigned int*)(p + k);
    t[2 * v]     = (unsigned short)(d & 0xffffu);
    t[2 * v + 1] = (unsigned short)(d >> 16);
  }
  return as_bf16x16(t);
}

// B-matrix (32x16, 16-bit) fragment: lane holds column (lane%16); element pair
// (2v,2v+1) sits at K = {2v | 16+2(v-4)} + 8*(lane/16).  p -> column start of K-contiguous storage.
__device__ __forceinline__ v16bf frag_b_from(const unsigned short* p, int half) {
  v16u t;
#pragma unroll
  for (int v = 0; v < 8; ++v) {
    int k = ((v < 4) ? 2 * v : 16 + 2 * (v - 4)) + half * 8;
    unsigned int d = *(const unsigned int*)(p + k);
    t[2 * v]     = (unsigned short)(d & 0xffffu);
    t[2 * v + 1] = (unsigned short)(d >> 16);
  }
  return as_bf16x16(t);
}

__device__ __forceinline__ v8f wmma_bf16(v16bf a, v16bf b, v8f c) {
  return __builtin_amdgcn_wmma_f32_16x16x32_bf16(false, a, false, b,
                                                 (short)0, c, false, false);
}

// ---------------------------------------------------------------------------
// Async global->LDS copy of 16 bytes (fallback: VGPR round trip)
// Builtin signature: (AS1 v4i* src, AS3 v4i* dst, imm offset, imm cpol)
// ---------------------------------------------------------------------------
__device__ __forceinline__ void copy16_to_lds(unsigned short* ldst,
                                              const unsigned short* gsrc) {
#ifdef HAVE_ASYNC
  __builtin_amdgcn_global_load_async_to_lds_b128(
      (__attribute__((address_space(1))) v4i*)(unsigned long long)gsrc,
      (__attribute__((address_space(3))) v4i*)
          (__attribute__((address_space(3))) void*)ldst,
      0, 0);
#else
  *(uint4*)ldst = *(const uint4*)gsrc;
#endif
}

__device__ __forceinline__ void lds_async_fence() {
#ifdef HAVE_ASYNC
  #if __has_builtin(__builtin_amdgcn_s_wait_asynccnt)
    __builtin_amdgcn_s_wait_asynccnt(0);
  #else
    asm volatile("s_wait_asynccnt 0" ::: "memory");
  #endif
#endif
}

// ---------------------------------------------------------------------------
// Tensor Data Mover: 2-D tile (bf16) global -> LDS with optional LDS row pad.
// D# layout per CDNA5 ISA 8.3/8.4. All dims in data_size (2B) units.
// ---------------------------------------------------------------------------
#ifdef HAVE_TDM
__device__ __forceinline__ void tdm_load_2d(unsigned lds_byte_off,
                                            const void* gaddr,
                                            unsigned tensor_d0, unsigned tensor_d1,
                                            unsigned stride0,
                                            unsigned tile_d0, unsigned tile_d1,
                                            unsigned pad_interval_code,
                                            unsigned pad_amount_code,
                                            int pad_en) {
  unsigned long long ga = (unsigned long long)gaddr;
  v4u g0;
  g0.x = 1u;                                    // count=1, user descriptor
  g0.y = lds_byte_off;                          // lds_addr
  g0.z = (unsigned)(ga & 0xffffffffu);          // global_addr[31:0]
  g0.w = (unsigned)((ga >> 32) & 0x01ffffffu) | (2u << 30);  // [56:32] | type=2
  unsigned d0 = (1u << 16);                     // data_size=1 -> 2 bytes
  if (pad_en)
    d0 |= (1u << 20) | (pad_interval_code << 22) | (pad_amount_code << 25);
  v8i g1;
  g1[0] = (int)d0;                              // wg_mask=0, flags
  g1[1] = (int)(tensor_d0 << 16);               // abar=0 | tensor_dim0[15:0]
  g1[2] = (int)((tensor_d0 >> 16) | (tensor_d1 << 16));
  g1[3] = (int)((tensor_d1 >> 16) | (tile_d0 << 16));
  g1[4] = (int)(tile_d1 & 0xffffu);             // tile_dim1 | tile_dim2=0
  g1[5] = (int)stride0;                         // tensor_dim0_stride[31:0]
  g1[6] = 0;                                    // stride hi=0 | dim1_stride lo=0
  g1[7] = 0;
  v4i gz = {0, 0, 0, 0};
#ifdef TDM_6ARG
  v8i gz8 = {0, 0, 0, 0, 0, 0, 0, 0};
  __builtin_amdgcn_tensor_load_to_lds(g0, g1, gz, gz, gz8, 0);
#else
  __builtin_amdgcn_tensor_load_to_lds(g0, g1, gz, gz, 0);
#endif
}
#endif

// ---------------------------------------------------------------------------
// Kernel 0: f32 -> bf16 cast
// ---------------------------------------------------------------------------
__global__ void cast_f32_bf16(const float* __restrict__ in,
                              unsigned short* __restrict__ out, int n) {
  int i = blockIdx.x * blockDim.x + threadIdx.x;
  int stride = gridDim.x * blockDim.x;
  for (; i < n; i += stride) out[i] = f2bf(in[i]);
}

// ---------------------------------------------------------------------------
// Kernel 1/5: bf16 WMMA GEMM   C[M,N] = A[M,K] * B[K,N] + bias[N]   (f32 out)
// BM=128, BN=128, BK=32; 256 threads = 8 waves (4x2); wave tile 32x64 = 2x4
// WMMA tiles (8 WMMA per K-step per wave).  A staged via async-to-LDS.
// ---------------------------------------------------------------------------
#define GBM 128
#define GBN 128
#define GBK 32
#define GLDA 40   // BK + 8 pad; 80B rows (16B aligned)
#define GLDB 40

template <int N, int K>
__launch_bounds__(256)
__global__ void gemm_bf16(const unsigned short* __restrict__ A,
                          const unsigned short* __restrict__ Bm,
                          const float* __restrict__ bias,
                          float* __restrict__ C) {
  __shared__ unsigned short As[GBM * GLDA];
  __shared__ unsigned short Bs[GBN * GLDB];   // stored transposed: [n][k]

  const int tid  = threadIdx.x;
  const int lane = tid & 31;
  const int wave = tid >> 5;
  const int wm   = wave >> 1;        // 0..3  (32-row strip)
  const int wn   = wave & 1;         // 0..1  (64-col strip)
  const int half = lane >> 4;
  const int l16  = lane & 15;
  const int rowBase = blockIdx.y * GBM;
  const int colBase = blockIdx.x * GBN;

  v8f acc[2][4];
#pragma unroll
  for (int i = 0; i < 2; ++i)
#pragma unroll
    for (int j = 0; j < 4; ++j)
      acc[i][j] = v8f{0.f, 0.f, 0.f, 0.f, 0.f, 0.f, 0.f, 0.f};

  for (int k0 = 0; k0 < K; k0 += GBK) {
    __syncthreads();
    // --- A tile 128x32: 16 elems/thread, async global->LDS b128 ---
    {
      int r = tid >> 2;            // 0..63
      int c = (tid & 3) * 8;       // 0,8,16,24
#pragma unroll
      for (int i = 0; i < 2; ++i) {
        int rr = r + i * 64;
        copy16_to_lds(&As[rr * GLDA + c],
                      A + (size_t)(rowBase + rr) * K + k0 + c);
      }
    }
    // --- B tile 32x128 transposed into Bs[n][k] (VGPR path) ---
    {
      int k = tid >> 4;            // 0..15
      int n = (tid & 15) * 8;      // 0..120
#pragma unroll
      for (int i = 0; i < 2; ++i) {
        int kk = k + i * 16;
        uint4 d = *(const uint4*)(Bm + (size_t)(k0 + kk) * N + colBase + n);
        const unsigned short* s = (const unsigned short*)&d;
#pragma unroll
        for (int j = 0; j < 8; ++j) Bs[(n + j) * GLDB + kk] = s[j];
      }
    }
    lds_async_fence();
    __syncthreads();

    v16bf af[2], bfr[4];
#pragma unroll
    for (int mt = 0; mt < 2; ++mt)
      af[mt] = frag_a_from(&As[(wm * 32 + mt * 16 + l16) * GLDA], half);
#pragma unroll
    for (int nt = 0; nt < 4; ++nt)
      bfr[nt] = frag_b_from(&Bs[(wn * 64 + nt * 16 + l16) * GLDB], half);
#pragma unroll
    for (int mt = 0; mt < 2; ++mt)
#pragma unroll
      for (int nt = 0; nt < 4; ++nt)
        acc[mt][nt] = wmma_bf16(af[mt], bfr[nt], acc[mt][nt]);
  }

  // epilogue: C element r -> row = r + 8*half, col = lane%16
#pragma unroll
  for (int nt = 0; nt < 4; ++nt) {
    int gcol = colBase + wn * 64 + nt * 16 + l16;
    float bv = bias[gcol];
#pragma unroll
    for (int mt = 0; mt < 2; ++mt)
#pragma unroll
      for (int r = 0; r < 8; ++r) {
        int grow = rowBase + wm * 32 + mt * 16 + r + 8 * half;
        C[(size_t)grow * N + gcol] = acc[mt][nt][r] + bv;
      }
  }
}

// ---------------------------------------------------------------------------
// Kernel 2: depthwise causal conv (DCONV=4) + RoPE split into bf16 Q/K/V
// Q: [B][H=16][S][128], K/V: [B][HKV=4][S][128]
// ---------------------------------------------------------------------------
__launch_bounds__(256)
__global__ void conv_rope(const float* __restrict__ qkv,      // [B*S][3072]
                          const float* __restrict__ conv_w,   // [3072][4]
                          const float* __restrict__ conv_b,   // [3072]
                          unsigned short* __restrict__ qb,
                          unsigned short* __restrict__ kb,
                          unsigned short* __restrict__ vb) {
  __shared__ float lds[3072];
  const int s = blockIdx.x;
  const int b = blockIdx.y;
  const int tid = threadIdx.x;

  for (int c = tid; c < 3072; c += 256) {
    float acc = conv_b[c];
#pragma unroll
    for (int k = 0; k < 4; ++k) {
      int ss = s - 3 + k;
      if (ss >= 0)
        acc += qkv[((size_t)b * 2048 + ss) * 3072 + c] * conv_w[c * 4 + k];
    }
    lds[c] = acc;
  }
  __syncthreads();

  const float fs = (float)s;
  const float L2B = 13.2877123795f;  // log2(10000)
  for (int c = tid; c < 2048; c += 256) {      // Q heads, rotary
    int h = c >> 7, d = c & 127, j = d & 63;
    float inv = exp2f(-(float)(2 * j) * (L2B / 128.f));
    float fr = fs * inv, cs = cosf(fr), sn = sinf(fr);
    float val = (d < 64) ? lds[c] * cs - lds[c + 64] * sn
                         : lds[c] * cs + lds[c - 64] * sn;
    qb[(((size_t)b * 16 + h) * 2048 + s) * 128 + d] = f2bf(val);
  }
  for (int c = tid; c < 512; c += 256) {       // K heads, rotary
    int h = c >> 7, d = c & 127, j = d & 63;
    int src = 2048 + c;
    float inv = exp2f(-(float)(2 * j) * (L2B / 128.f));
    float fr = fs * inv, cs = cosf(fr), sn = sinf(fr);
    float val = (d < 64) ? lds[src] * cs - lds[src + 64] * sn
                         : lds[src] * cs + lds[src - 64] * sn;
    kb[(((size_t)b * 4 + h) * 2048 + s) * 128 + d] = f2bf(val);
  }
  for (int c = tid; c < 512; c += 256) {       // V heads
    vb[(((size_t)b * 4 + (c >> 7)) * 2048 + s) * 128 + (c & 127)] =
        f2bf(lds[2560 + c]);
  }
}

// ---------------------------------------------------------------------------
// Kernel 3: causal flash attention with bf16 WMMA (QK^T and PV)
// Block = 128 threads (4 waves), 64 q-rows per block (16 per wave);
// K tile staged by the Tensor Data Mover (wave 0), V transposed by VALU.
// ---------------------------------------------------------------------------
#define ALDK 136   // 128+8 shorts -> 272B rows (16B aligned; TDM pad 4DW/64DW)
#define ALDV 72    // 64+8 shorts  -> 144B rows (16B aligned)

__launch_bounds__(128)
__global__ void flash_attn(const unsigned short* __restrict__ qg,
                           const unsigned short* __restrict__ kg_,
                           const unsigned short* __restrict__ vg_,
                           unsigned short* __restrict__ ctx) {
  __shared__ unsigned short Ks[64 * ALDK];
  __shared__ unsigned short Vs[128 * ALDV];
  __shared__ float          sc[4][16][64];
  __shared__ unsigned short ps[4][16][64];
  __shared__ float mrow[4][16], lrow[4][16], crow[4][16];

  const int qblk = blockIdx.x, h = blockIdx.y, b = blockIdx.z;
  const int hkv  = h >> 2;                    // H/HKV = 4
  const int tid  = threadIdx.x;
  const int lane = tid & 31;
  const int wave = tid >> 5;
  const int half = lane >> 4;
  const int l16  = lane & 15;
  const int q0   = qblk * 64 + wave * 16;

  // Q fragments for this wave's 16 rows: 4 chunks of K=32 over D=128
  const unsigned short* qrow =
      qg + (((size_t)b * 16 + h) * 2048 + q0 + l16) * 128;
  v16bf qf[4];
#pragma unroll
  for (int c = 0; c < 4; ++c) qf[c] = frag_a_from(qrow + c * 32, half);

  v8f o[8];
#pragma unroll
  for (int n = 0; n < 8; ++n) o[n] = v8f{0.f,0.f,0.f,0.f,0.f,0.f,0.f,0.f};
  if (lane < 16) { mrow[wave][lane] = -3.0e38f; lrow[wave][lane] = 0.f; }

  const float scale = 0.08838834764831845f;   // 1/sqrt(128)
  const unsigned short* kg = kg_ + ((size_t)b * 4 + hkv) * 2048 * 128;
  const unsigned short* vg = vg_ + ((size_t)b * 4 + hkv) * 2048 * 128;

#ifdef HAVE_TDM
  const unsigned ks_off = (unsigned)(unsigned long long)
      (__attribute__((address_space(3))) void*)&Ks[0];
#endif

  for (int j = 0; j <= qblk; ++j) {
    const int kv0 = j * 64;
    __syncthreads();
    // --- stage K tile [64][128] ---
#ifdef HAVE_TDM
    if (wave == 0) {
      // 64 rows x 128 bf16, row stride 128; pad LDS by 4 DWORDs every 64 DWORDs
      tdm_load_2d(ks_off, kg + (size_t)kv0 * 128,
                  /*tensor_d0=*/128, /*tensor_d1=*/64, /*stride0=*/128,
                  /*tile_d0=*/128, /*tile_d1=*/64,
                  /*pad_interval(64DW)=*/5, /*pad_amount(4DW)=*/3, 1);
      __builtin_amdgcn_s_wait_tensorcnt(0);
    }
#else
    {
      int r = tid >> 4;            // 0..7
      int c = (tid & 15) * 8;      // 0..120
#pragma unroll
      for (int i = 0; i < 8; ++i) {
        int rr = r + i * 8;
        copy16_to_lds(&Ks[rr * ALDK + c], kg + (size_t)(kv0 + rr) * 128 + c);
      }
    }
#endif
    // --- stage V tile transposed [128][64] ---
    {
      int r = tid >> 4;
      int c = (tid & 15) * 8;
#pragma unroll
      for (int i = 0; i < 8; ++i) {
        int rr = r + i * 8;
        uint4 dv = *(const uint4*)(vg + (size_t)(kv0 + rr) * 128 + c);
        const unsigned short* sp = (const unsigned short*)&dv;
#pragma unroll
        for (int jj = 0; jj < 8; ++jj) Vs[(c + jj) * ALDV + rr] = sp[jj];
      }
    }
#ifndef HAVE_TDM
    lds_async_fence();
#endif
    __syncthreads();

    // --- scores: 16x64 = 4 n-tiles, K-dim = D = 4 chunks of 32 ---
    v8f sa[4];
#pragma unroll
    for (int nt = 0; nt < 4; ++nt) sa[nt] = v8f{0.f,0.f,0.f,0.f,0.f,0.f,0.f,0.f};
#pragma unroll
    for (int nt = 0; nt < 4; ++nt)
#pragma unroll
      for (int c = 0; c < 4; ++c) {
        v16bf kf = frag_b_from(&Ks[(nt * 16 + l16) * ALDK + c * 32], half);
        sa[nt] = wmma_bf16(qf[c], kf, sa[nt]);
      }
#pragma unroll
    for (int nt = 0; nt < 4; ++nt)
#pragma unroll
      for (int r = 0; r < 8; ++r)
        sc[wave][r + 8 * half][nt * 16 + l16] = sa[nt][r];

    // --- online softmax, one row per lane (lanes 0..15) ---
    if (lane < 16) {
      int row  = lane;
      int kmax = (q0 + row) - kv0;            // valid keys: kk <= kmax (>=0)
      float mold = mrow[wave][row];
      float mnew = mold;
      for (int kk = 0; kk < 64; ++kk) {
        float v = sc[wave][row][kk] * scale;
        if (kk <= kmax && v > mnew) mnew = v;
      }
      float corr = __expf(mold - mnew);
      float sum = 0.f;
      for (int kk = 0; kk < 64; ++kk) {
        float p = (kk <= kmax) ? __expf(sc[wave][row][kk] * scale - mnew) : 0.f;
        sum += p;
        ps[wave][row][kk] = f2bf(p);
      }
      mrow[wave][row] = mnew;
      lrow[wave][row] = lrow[wave][row] * corr + sum;
      crow[wave][row] = corr;
    }
    // wave lockstep: all 32 lanes see ps/crow after the divergent block
#pragma unroll
    for (int n = 0; n < 8; ++n)
#pragma unroll
      for (int r = 0; r < 8; ++r)
        o[n][r] *= crow[wave][r + 8 * half];

    // --- PV: P(16x64) * V(64x128); K-dim = 2 chunks of 32 ---
    v16bf pf[2];
#pragma unroll
    for (int c = 0; c < 2; ++c)
      pf[c] = frag_a_from(&ps[wave][l16][c * 32], half);
#pragma unroll
    for (int n = 0; n < 8; ++n)
#pragma unroll
      for (int c = 0; c < 2; ++c) {
        v16bf vf = frag_b_from(&Vs[(n * 16 + l16) * ALDV + c * 32], half);
        o[n] = wmma_bf16(pf[c], vf, o[n]);
      }
  }

  // --- normalize and store ctx as bf16 [b][s][h*128+d] ---
#pragma unroll
  for (int n = 0; n < 8; ++n)
#pragma unroll
    for (int r = 0; r < 8; ++r) {
      int row = r + 8 * half;
      float linv = 1.0f / lrow[wave][row];
      int s = q0 + row;
      int d = n * 16 + l16;
      ctx[((size_t)b * 2048 + s) * 2048 + h * 128 + d] = f2bf(o[n][r] * linv);
    }
}

// ---------------------------------------------------------------------------
// Launch
// ---------------------------------------------------------------------------
extern "C" void kernel_launch(void* const* d_in, const int* in_sizes, int n_in,
                              void* d_out, int out_size, void* d_ws, size_t ws_size,
                              hipStream_t stream) {
  (void)in_sizes; (void)n_in; (void)out_size; (void)ws_size;
  const float* x      = (const float*)d_in[0];
  const float* W_in   = (const float*)d_in[1];
  const float* b_in   = (const float*)d_in[2];
  const float* conv_w = (const float*)d_in[3];
  const float* conv_b = (const float*)d_in[4];
  const float* W_out  = (const float*)d_in[5];
  const float* b_out  = (const float*)d_in[6];
  float* out = (float*)d_out;

  char* ws = (char*)d_ws;
  unsigned short* xb    = (unsigned short*)(ws + 0);           // 16 MB
  unsigned short* winb  = (unsigned short*)(ws + 16777216);    // 12 MB
  unsigned short* woutb = (unsigned short*)(ws + 29360128);    //  8 MB
  float*          qkvf  = (float*)         (ws + 37748736);    // 48 MB
  unsigned short* qb    = (unsigned short*)(ws + 88080384);    // 16 MB
  unsigned short* kb    = (unsigned short*)(ws + 104857600);   //  4 MB
  unsigned short* vb    = (unsigned short*)(ws + 109051904);   //  4 MB
  unsigned short* ctxb  = (unsigned short*)(ws + 113246208);   // 16 MB

  cast_f32_bf16<<<2048, 256, 0, stream>>>(x, xb, 2 * 2048 * 2048);
  cast_f32_bf16<<<2048, 256, 0, stream>>>(W_in, winb, 2048 * 3072);
  cast_f32_bf16<<<2048, 256, 0, stream>>>(W_out, woutb, 2048 * 2048);

  // qkv = x @ W_in + b_in   (M=4096, N=3072, K=2048)
  gemm_bf16<3072, 2048><<<dim3(3072 / GBN, 4096 / GBM), 256, 0, stream>>>(
      xb, winb, b_in, qkvf);

  conv_rope<<<dim3(2048, 2), 256, 0, stream>>>(qkvf, conv_w, conv_b, qb, kb, vb);

  flash_attn<<<dim3(2048 / 64, 16, 2), 128, 0, stream>>>(qb, kb, vb, ctxb);

  // out = ctx @ W_out + b_out   (M=4096, N=2048, K=2048)
  gemm_bf16<2048, 2048><<<dim3(2048 / GBN, 4096 / GBM), 256, 0, stream>>>(
      ctxb, woutb, b_out, out);
}
